// GAT_37426345017680
// MI455X (gfx1250) — matrix operand order repaired and checked by reading, not compiled, not run on hardware
//
#include <hip/hip_runtime.h>

// ---------------- problem constants ----------------
#define GN 100000      // nodes
#define GE 3200000     // edges
#define GD 128         // input dim
#define GH1 16         // conv1 out channels
#define NTILES (GN / 16)   // 6250 exact

typedef __attribute__((ext_vector_type(16))) _Float16 v16h;
typedef __attribute__((ext_vector_type(8)))  float    v8f;

// order-preserving float<->uint key for atomicMax-based segment max
__device__ __forceinline__ unsigned fkey(float f) {
    unsigned u = __float_as_uint(f);
    return (u & 0x80000000u) ? ~u : (u | 0x80000000u);
}
__device__ __forceinline__ float fdec(unsigned k) {
    return (k & 0x80000000u) ? __uint_as_float(k & 0x7FFFFFFFu)
                             : __uint_as_float(~k);
}

// ---------------- zero the atomic accumulators ----------------
__global__ void gat_zero(unsigned* p, int n) {
    for (int i = blockIdx.x * blockDim.x + threadIdx.x; i < n;
         i += gridDim.x * blockDim.x)
        p[i] = 0u;
}

// ---------------- prep: fold vectors + pre-swizzled f16 B operands ----------------
// bB1/bB2 layout: [kk(4)][lane(32)][idx(16)] halfs, in a contiguous-k order (not the ISA
// table order), shared with the A loads, for V_WMMA_F32_16X16X32_F16:
//   element (K,N): lane = 16*(K%32/16) + N, idx = K%16, kk = K/32
__global__ void gat_prep(const float* __restrict__ W1s, const float* __restrict__ W1d,
                         const float* __restrict__ a1s, const float* __restrict__ a1d,
                         _Float16* __restrict__ bB1, _Float16* __restrict__ bB2) {
    __shared__ float ws_[GD], wd_[GD];
    int t = threadIdx.x;                 // 128 threads
    float ss = 0.f, dd = 0.f;
    #pragma unroll
    for (int n = 0; n < GH1; ++n) {
        ss += W1s[t * GH1 + n] * a1s[n];
        dd += W1d[t * GH1 + n] * a1d[n];
    }
    ws_[t] = ss; wd_[t] = dd;
    __syncthreads();
    for (int i = t; i < 4 * 32 * 16; i += 128) {
        int idx  = i & 15;
        int lane = (i >> 4) & 31;
        int kk   = i >> 9;
        int K    = kk * 32 + (lane >> 4) * 16 + idx;
        int Ncol = lane & 15;
        bB1[i] = (_Float16)W1s[K * GH1 + Ncol];
        float v = (Ncol == 0) ? ws_[K] : ((Ncol == 1) ? wd_[K] : 0.f);
        bB2[i] = (_Float16)v;
    }
}

// ---------------- WMMA projection: h1 = x@W1_src, a_s = x@(W1s a1s), a_d = x@(W1d a1d) ----------------
__global__ void gat_proj_wmma(const float* __restrict__ x,
                              const _Float16* __restrict__ bB1,
                              const _Float16* __restrict__ bB2,
                              float* __restrict__ h1,
                              float* __restrict__ a_s,
                              float* __restrict__ a_d) {
    const int lane = threadIdx.x & 31;
    const int wave = threadIdx.x >> 5;              // 8 waves / block
    const int tile = blockIdx.x * 8 + wave;         // 16 nodes per tile
    if (tile >= NTILES) return;                     // uniform per wave -> EXEC stays full
    const int row0 = tile * 16;
    const int m  = lane & 15;                       // A: M = lane%16 ; C: N = lane%16
    const int hh = lane >> 4;

    v8f c1 = {};                                    // h tile accumulator
    v8f c2 = {};                                    // fused [a_s | a_d | 0...] accumulator
    #pragma unroll
    for (int kk = 0; kk < 4; ++kk) {
        // A operand: lane holds row m, K = kk*32 + 16*hh + {0..15} (matches bB packing)
        const float* xr = x + (size_t)(row0 + m) * GD + kk * 32 + 16 * hh;
        v16h a;
        #pragma unroll
        for (int j = 0; j < 8; ++j) a[j]     = (_Float16)xr[j];
        #pragma unroll
        for (int j = 0; j < 8; ++j) a[8 + j] = (_Float16)xr[8 + j];

        const v16h b1 = *(const v16h*)(bB1 + ((size_t)kk * 32 + lane) * 16);
        const v16h b2 = *(const v16h*)(bB2 + ((size_t)kk * 32 + lane) * 16);
        c1 = __builtin_amdgcn_wmma_f32_16x16x32_f16(false, a, false, b1, (short)0, c1, false, false);
        c2 = __builtin_amdgcn_wmma_f32_16x16x32_f16(false, a, false, b2, (short)0, c2, false, false);
    }
    // C layout: vgpr v, lane: M = v + 8*hh, N = m
    #pragma unroll
    for (int v = 0; v < 8; ++v)
        h1[(size_t)(row0 + v + 8 * hh) * GH1 + m] = c1[v];
    if (m == 0) {
        #pragma unroll
        for (int v = 0; v < 8; ++v) a_s[row0 + v + 8 * hh] = c2[v];
    } else if (m == 1) {
        #pragma unroll
        for (int v = 0; v < 8; ++v) a_d[row0 + v + 8 * hh] = c2[v];
    }
}

// ---------------- edge pass 1: leaky-relu score + segment max ----------------
__global__ void gat_edge_score(const int* __restrict__ ei,
                               const float* __restrict__ as, const float* __restrict__ ad,
                               float* __restrict__ ebuf, unsigned* __restrict__ mkey) {
    for (int e = blockIdx.x * blockDim.x + threadIdx.x; e < GE;
         e += gridDim.x * blockDim.x) {
        int s = ei[e], d = ei[GE + e];
        float v = as[s] + ad[d];
        v = v > 0.f ? v : 0.2f * v;
        ebuf[e] = v;
        atomicMax(&mkey[d], fkey(v));
    }
}

// ---------------- edge pass 2: exp + segment sum ----------------
__global__ void gat_edge_exp(const int* __restrict__ ei, float* __restrict__ ebuf,
                             const unsigned* __restrict__ mkey, float* __restrict__ z) {
    for (int e = blockIdx.x * blockDim.x + threadIdx.x; e < GE;
         e += gridDim.x * blockDim.x) {
        int d = ei[GE + e];
        float p = __expf(ebuf[e] - fdec(mkey[d]));
        ebuf[e] = p;
        atomicAdd(&z[d], p);
    }
}

// ---------------- edge pass 3: alpha-weighted scatter-add of 16-float rows ----------------
__global__ void gat_edge_agg16(const int* __restrict__ ei, const float* __restrict__ ebuf,
                               const float* __restrict__ z, const float* __restrict__ h1,
                               float* __restrict__ agg) {
    for (int e = blockIdx.x * blockDim.x + threadIdx.x; e < GE;
         e += gridDim.x * blockDim.x) {
        int s = ei[e], d = ei[GE + e];
        float alpha = ebuf[e] / (z[d] + 1e-16f);
        const float4* hs = (const float4*)(h1 + (size_t)s * GH1);
        float* o = agg + (size_t)d * GH1;
        #pragma unroll
        for (int q = 0; q < 4; ++q) {
            float4 hv = hs[q];
            atomicAdd(&o[q * 4 + 0], alpha * hv.x);
            atomicAdd(&o[q * 4 + 1], alpha * hv.y);
            atomicAdd(&o[q * 4 + 2], alpha * hv.z);
            atomicAdd(&o[q * 4 + 3], alpha * hv.w);
        }
    }
}

// ---------------- layer-2 node pass: relu + 16-dot + attention scalars ----------------
__global__ void gat_l2_node(const float* __restrict__ agg1, const float* __restrict__ b1,
                            const float* __restrict__ W2, const float* __restrict__ a2sw,
                            const float* __restrict__ a2dw,
                            float* __restrict__ h2, float* __restrict__ a2s,
                            float* __restrict__ a2d) {
    int i = blockIdx.x * blockDim.x + threadIdx.x;
    if (i >= GN) return;
    float acc = 0.f;
    #pragma unroll
    for (int n = 0; n < GH1; ++n) {
        float v = agg1[(size_t)i * GH1 + n] + b1[n];
        v = fmaxf(v, 0.f);
        acc += v * W2[n];
    }
    h2[i]  = acc;
    a2s[i] = acc * a2sw[0];
    a2d[i] = acc * a2dw[0];
}

// ---------------- layer-2 scalar aggregation ----------------
__global__ void gat_edge_agg1(const int* __restrict__ ei, const float* __restrict__ ebuf,
                              const float* __restrict__ z, const float* __restrict__ h2,
                              float* __restrict__ agg) {
    for (int e = blockIdx.x * blockDim.x + threadIdx.x; e < GE;
         e += gridDim.x * blockDim.x) {
        int s = ei[e], d = ei[GE + e];
        float alpha = ebuf[e] / (z[d] + 1e-16f);
        atomicAdd(&agg[d], alpha * h2[s]);
    }
}

__global__ void gat_final(const float* __restrict__ agg2, const float* __restrict__ b2,
                          float* __restrict__ out) {
    int i = blockIdx.x * blockDim.x + threadIdx.x;
    if (i < GN) out[i] = agg2[i] + b2[0];
}

// ---------------- launcher ----------------
static inline size_t alignUp(size_t v, size_t a) { return (v + a - 1) & ~(a - 1); }

extern "C" void kernel_launch(void* const* d_in, const int* in_sizes, int n_in,
                              void* d_out, int out_size, void* d_ws, size_t ws_size,
                              hipStream_t stream) {
    const float* x     = (const float*)d_in[0];
    const int*   ei    = (const int*)d_in[1];
    const float* W1s   = (const float*)d_in[2];
    const float* W1d   = (const float*)d_in[3];
    const float* a1s   = (const float*)d_in[4];
    const float* a1d   = (const float*)d_in[5];
    const float* b1    = (const float*)d_in[6];
    const float* W2    = (const float*)d_in[7];
    const float* a2sw  = (const float*)d_in[8];
    const float* a2dw  = (const float*)d_in[9];
    const float* b2    = (const float*)d_in[10];
    float* out = (float*)d_out;

    // ---- workspace carve-up ----
    char* ws = (char*)d_ws;
    size_t off = 0;
    auto take = [&](size_t bytes) { size_t o = off; off = alignUp(off + bytes, 256); return o; };

    float*    h1    = (float*)(ws + take((size_t)GN * GH1 * 4));
    float*    a_s   = (float*)(ws + take((size_t)GN * 4));
    float*    a_d   = (float*)(ws + take((size_t)GN * 4));
    float*    h2    = (float*)(ws + take((size_t)GN * 4));
    float*    a2s   = (float*)(ws + take((size_t)GN * 4));
    float*    a2d   = (float*)(ws + take((size_t)GN * 4));
    float*    ebuf  = (float*)(ws + take((size_t)GE * 4));
    // contiguous zeroed accumulator block
    size_t zero_off = off;
    float*    agg1  = (float*)(ws + take((size_t)GN * GH1 * 4));
    unsigned* mkey  = (unsigned*)(ws + take((size_t)GN * 4));
    float*    zsum  = (float*)(ws + take((size_t)GN * 4));
    unsigned* mkey2 = (unsigned*)(ws + take((size_t)GN * 4));
    float*    zsum2 = (float*)(ws + take((size_t)GN * 4));
    float*    agg2  = (float*)(ws + take((size_t)GN * 4));
    size_t zero_bytes = off - zero_off;
    _Float16* bB1   = (_Float16*)(ws + take(4 * 32 * 16 * 2));
    _Float16* bB2   = (_Float16*)(ws + take(4 * 32 * 16 * 2));

    const int zn = (int)(zero_bytes / 4);
    gat_zero<<<2048, 256, 0, stream>>>((unsigned*)(ws + zero_off), zn);
    gat_prep<<<1, 128, 0, stream>>>(W1s, W1d, a1s, a1d, bB1, bB2);

    const int wmmaBlocks = (NTILES + 7) / 8;        // 8 waves (tiles) per 256-thread block
    gat_proj_wmma<<<wmmaBlocks, 256, 0, stream>>>(x, bB1, bB2, h1, a_s, a_d);

    gat_edge_score<<<2048, 256, 0, stream>>>(ei, a_s, a_d, ebuf, mkey);
    gat_edge_exp<<<2048, 256, 0, stream>>>(ei, ebuf, mkey, zsum);
    gat_edge_agg16<<<2048, 256, 0, stream>>>(ei, ebuf, zsum, h1, agg1);

    gat_l2_node<<<(GN + 255) / 256, 256, 0, stream>>>(agg1, b1, W2, a2sw, a2dw, h2, a2s, a2d);

    gat_edge_score<<<2048, 256, 0, stream>>>(ei, a2s, a2d, ebuf, mkey2);
    gat_edge_exp<<<2048, 256, 0, stream>>>(ei, ebuf, mkey2, zsum2);
    gat_edge_agg1<<<2048, 256, 0, stream>>>(ei, ebuf, zsum2, h2, agg2);

    gat_final<<<(GN + 255) / 256, 256, 0, stream>>>(agg2, b2, out);
}
